// GraphConvNet_28406913696300
// MI455X (gfx1250) — matrix operand order repaired and checked by reading, not compiled, not run on hardware
//
#include <hip/hip_runtime.h>
#include <hip/hip_bf16.h>

typedef __attribute__((ext_vector_type(16))) __bf16 v16bf;
typedef __attribute__((ext_vector_type(8)))  float  v8f;

#define N_NODES 50000
#define N_EDGES 800000
#define IN_F    7
#define G_DIM   16
#define LATENT  64
#define HIDDEN  128

// ---------- helpers ----------

__device__ __forceinline__ unsigned short f2bf(float f) {
  union { float f; unsigned int u; } v; v.f = f;
  unsigned int u = v.u;
  u += 0x7FFFu + ((u >> 16) & 1u);   // round-to-nearest-even
  return (unsigned short)(u >> 16);
}

__device__ __forceinline__ float gelu_tanh(float x) {
  const float c0 = 0.7978845608028654f;   // sqrt(2/pi)
  const float c1 = 0.044715f;
  return 0.5f * x * (1.0f + tanhf(c0 * (x + c1 * x * x * x)));
}

union FragU { v16bf v; uint4 q[2]; };

// A-matrix 16x32 bf16 fragment from row-major LDS tile [rows x Kpad].
// lane<16: row = mt*16+lane, K = kt*32 + {0..7, 16..23}
// lane>=16: row = mt*16+lane-16, K = kt*32 + {8..15, 24..31}
__device__ __forceinline__ v16bf load_a_frag(const unsigned short* lds, int Kpad,
                                             int mt, int kt, int lane) {
  int row = mt * 16 + (lane & 15);
  const unsigned short* p = lds + row * Kpad + kt * 32 + ((lane & 16) ? 8 : 0);
  FragU f;
  f.q[0] = *(const uint4*)(p);        // elems 0..7
  f.q[1] = *(const uint4*)(p + 16);   // elems 8..15 (K += 16)
  return f.v;
}

// B-matrix 32x16 bf16 fragment from pre-packed global buffer:
// packed index = ((kt*nct + ct)*32 + lane)*16 + elem, lane-contiguous 32B.
__device__ __forceinline__ v16bf load_b_frag(const unsigned short* pw, int nct,
                                             int kt, int ct, int lane) {
  const uint4* p = (const uint4*)(pw + (size_t)(((kt * nct + ct) * 32) + lane) * 16);
  FragU f;
  f.q[0] = p[0];
  f.q[1] = p[1];
  return f.v;
}

__device__ __forceinline__ v8f wmma_bf16(v16bf a, v16bf b, v8f c) {
  return __builtin_amdgcn_wmma_f32_16x16x32_bf16(false, a, false, b,
                                                 (short)0, c, false, false);
}

// ---------- weight pre-pack: f32 [Kact x Ncols] -> bf16 B-fragment order ----------
__global__ void pack_w_kernel(const float* __restrict__ w, unsigned short* __restrict__ out,
                              int Kact, int Kpad, int Ncols) {
  int total = Kpad * Ncols;
  int i = blockIdx.x * 256 + threadIdx.x;
  if (i >= total) return;
  int j    = i & 15;
  int lane = (i >> 4) & 31;
  int rest = i >> 9;
  int nct  = Ncols >> 4;
  int ct   = rest % nct;
  int kt   = rest / nct;
  int n = ct * 16 + (lane & 15);
  int k = kt * 32 + ((j & 8) ? 16 : 0) + ((lane & 16) ? 8 : 0) + (j & 7);
  float v = (k < Kact) ? w[(size_t)k * Ncols + n] : 0.0f;
  out[i] = f2bf(v);
}

// ---------- small elementwise kernels ----------
__global__ void zero_f32_kernel(float* __restrict__ p, int n) {
  int i = blockIdx.x * 256 + threadIdx.x;
  if (i < n) p[i] = 0.0f;
}

__global__ void embed_kernel(const float* __restrict__ nodes,
                             const float* __restrict__ w,
                             const float* __restrict__ b,
                             float* __restrict__ h) {
  int i = blockIdx.x * 256 + threadIdx.x;
  if (i >= N_NODES * LATENT) return;
  int node = i / LATENT, col = i - node * LATENT;
  float acc = b[col];
#pragma unroll
  for (int k = 0; k < IN_F; ++k)
    acc += nodes[node * IN_F + k] * w[k * LATENT + col];
  h[i] = acc;
}

__global__ void decode_kernel(const float* __restrict__ h,
                              const float* __restrict__ w,
                              const float* __restrict__ b,
                              float* __restrict__ out) {
  int i = blockIdx.x * 256 + threadIdx.x;
  if (i >= N_NODES * IN_F) return;
  int node = i / IN_F, f = i - node * IN_F;
  float acc = b[f];
  for (int c = 0; c < LATENT; ++c)
    acc += h[(size_t)node * LATENT + c] * w[c * IN_F + f];
  out[i] = acc;
}

// ---------- edge MLP: 32 edges per 256-thread block ----------
__global__ __launch_bounds__(256, 2) void edge_mlp_kernel(
    const float* __restrict__ h,
    unsigned short* __restrict__ e_buf,          // [E x 64] bf16, in/out
    const int* __restrict__ senders,
    const int* __restrict__ receivers,
    const float* __restrict__ g,
    const unsigned short* __restrict__ w1p,      // packed [Kpad x 128]
    const float* __restrict__ b1,                // [128]
    const unsigned short* __restrict__ w2p,      // packed [128 x 64]
    const float* __restrict__ b2,                // [64]
    float* __restrict__ recv,                    // [N x 64] f32, atomic acc
    int has_e, int Kpad) {
  __shared__ __align__(16) unsigned short s_ein[32 * 224];
  __shared__ __align__(16) unsigned short s_hid[32 * HIDDEN];
  __shared__ int s_snd[32];
  __shared__ int s_rcv[32];

  const int tid = threadIdx.x;
  const int e0 = blockIdx.x * 32;
  if (tid < 32) {
    s_snd[tid] = senders[e0 + tid];
    s_rcv[tid] = receivers[e0 + tid];
  }
  __syncthreads();

  // build e_in = [e? | s | r | g | pad] in bf16
  const int sc = has_e ? 64 : 0;
  const int rc = sc + 64;
  const int gc = rc + 64;
  const int pc = gc + G_DIM;
  for (int i = tid; i < 32 * Kpad; i += 256) {
    int row = i / Kpad;
    int col = i - row * Kpad;
    unsigned short v;
    if (col < sc)      v = e_buf[(size_t)(e0 + row) * LATENT + col];
    else if (col < rc) v = f2bf(h[(size_t)s_snd[row] * LATENT + (col - sc)]);
    else if (col < gc) v = f2bf(h[(size_t)s_rcv[row] * LATENT + (col - rc)]);
    else if (col < pc) v = f2bf(g[col - gc]);
    else               v = 0;
    s_ein[row * Kpad + col] = v;
  }
  __syncthreads();

  const int lane = tid & 31;
  const int wave = tid >> 5;
  const int nkt = Kpad >> 5;

  // phase 1: [32 x Kpad] @ [Kpad x 128] -> gelu -> LDS bf16
  for (int mt = 0; mt < 2; ++mt) {
    const int ct = wave;                 // 8 waves cover 128 cols
    v8f acc = {};
    for (int kt = 0; kt < nkt; ++kt) {
      v16bf a = load_a_frag(s_ein, Kpad, mt, kt, lane);
      v16bf b = load_b_frag(w1p, 8, kt, ct, lane);
      acc = wmma_bf16(a, b, acc);
    }
    const int col = ct * 16 + (lane & 15);
    const float bias = b1[col];
    const int rbase = mt * 16 + ((lane & 16) ? 8 : 0);
#pragma unroll
    for (int r = 0; r < 8; ++r)
      s_hid[(rbase + r) * HIDDEN + col] = f2bf(gelu_tanh(acc[r] + bias));
  }
  __syncthreads();

  // phase 2: [32 x 128] @ [128 x 64]; store bf16 e, atomic-add f32 into recv
  {
    const int mt = wave >> 2;
    const int ct = wave & 3;
    v8f acc = {};
#pragma unroll
    for (int kt = 0; kt < 4; ++kt) {
      v16bf a = load_a_frag(s_hid, HIDDEN, mt, kt, lane);
      v16bf b = load_b_frag(w2p, 4, kt, ct, lane);
      acc = wmma_bf16(a, b, acc);
    }
    const int col = ct * 16 + (lane & 15);
    const float bias = b2[col];
    const int rbase = mt * 16 + ((lane & 16) ? 8 : 0);
#pragma unroll
    for (int r = 0; r < 8; ++r) {
      const int row = rbase + r;
      const float v = acc[r] + bias;
      e_buf[(size_t)(e0 + row) * LATENT + col] = f2bf(v);
      unsafeAtomicAdd(&recv[(size_t)s_rcv[row] * LATENT + col], v);
    }
  }
}

// ---------- node MLP + residual + LayerNorm: 32 nodes per block ----------
__global__ __launch_bounds__(256, 2) void node_mlp_kernel(
    float* __restrict__ h,
    const float* __restrict__ recv,
    const float* __restrict__ g,
    const unsigned short* __restrict__ w1p,      // packed [160 x 128]
    const float* __restrict__ b1,
    const unsigned short* __restrict__ w2p,      // packed [128 x 64]
    const float* __restrict__ b2,
    const float* __restrict__ ln_w,
    const float* __restrict__ ln_b) {
  __shared__ __align__(16) unsigned short s_nin[32 * 160];
  __shared__ __align__(16) unsigned short s_hid[32 * HIDDEN];
  __shared__ float s_x[32 * LATENT];
  const int Kpad = 160;
  const int tid = threadIdx.x;
  const int n0 = blockIdx.x * 32;

  for (int i = tid; i < 32 * Kpad; i += 256) {
    int row = i / Kpad;
    int col = i - row * Kpad;
    int node = n0 + row;
    float v = 0.0f;
    if (node < N_NODES) {
      if (col < 64)        v = h[(size_t)node * LATENT + col];
      else if (col < 128)  v = recv[(size_t)node * LATENT + (col - 64)];
      else if (col < 144)  v = g[col - 128];
    }
    s_nin[i] = f2bf(v);
  }
  __syncthreads();

  const int lane = tid & 31;
  const int wave = tid >> 5;

  for (int mt = 0; mt < 2; ++mt) {
    const int ct = wave;
    v8f acc = {};
#pragma unroll
    for (int kt = 0; kt < 5; ++kt) {
      v16bf a = load_a_frag(s_nin, Kpad, mt, kt, lane);
      v16bf b = load_b_frag(w1p, 8, kt, ct, lane);
      acc = wmma_bf16(a, b, acc);
    }
    const int col = ct * 16 + (lane & 15);
    const float bias = b1[col];
    const int rbase = mt * 16 + ((lane & 16) ? 8 : 0);
#pragma unroll
    for (int r = 0; r < 8; ++r)
      s_hid[(rbase + r) * HIDDEN + col] = f2bf(gelu_tanh(acc[r] + bias));
  }
  __syncthreads();

  {
    const int mt = wave >> 2;
    const int ct = wave & 3;
    v8f acc = {};
#pragma unroll
    for (int kt = 0; kt < 4; ++kt) {
      v16bf a = load_a_frag(s_hid, HIDDEN, mt, kt, lane);
      v16bf b = load_b_frag(w2p, 4, kt, ct, lane);
      acc = wmma_bf16(a, b, acc);
    }
    const int col = ct * 16 + (lane & 15);
    const float bias = b2[col];
    const int rbase = mt * 16 + ((lane & 16) ? 8 : 0);
#pragma unroll
    for (int r = 0; r < 8; ++r) {
      const int row = rbase + r;
      const int node = n0 + row;
      float x = acc[r] + bias;                           // upd
      if (node < N_NODES) x += h[(size_t)node * LATENT + col];  // residual
      s_x[row * LATENT + col] = x;
    }
  }
  __syncthreads();

  // LayerNorm over 64 latent dims, one thread per row
  if (tid < 32) {
    const int node = n0 + tid;
    if (node < N_NODES) {
      float mu = 0.0f, m2 = 0.0f;
      for (int c = 0; c < LATENT; ++c) {
        float x = s_x[tid * LATENT + c];
        mu += x; m2 += x * x;
      }
      mu *= (1.0f / LATENT);
      float var = m2 * (1.0f / LATENT) - mu * mu;
      float rs = rsqrtf(var + 1e-6f);
      for (int c = 0; c < LATENT; ++c) {
        float x = s_x[tid * LATENT + c];
        h[(size_t)node * LATENT + c] = (x - mu) * rs * ln_w[c] + ln_b[c];
      }
    }
  }
}

// ---------- launcher ----------
extern "C" void kernel_launch(void* const* d_in, const int* in_sizes, int n_in,
                              void* d_out, int out_size, void* d_ws, size_t ws_size,
                              hipStream_t stream) {
  const float* nodes      = (const float*)d_in[0];
  const int*   senders    = (const int*)d_in[1];
  const int*   receivers  = (const int*)d_in[2];
  const float* g          = (const float*)d_in[3];
  const float* embed_w    = (const float*)d_in[4];
  const float* embed_b    = (const float*)d_in[5];
  const float* e_w1_first = (const float*)d_in[6];
  const float* e_w1_rest  = (const float*)d_in[7];
  const float* e_b1       = (const float*)d_in[8];
  const float* e_w2       = (const float*)d_in[9];
  const float* e_b2       = (const float*)d_in[10];
  const float* n_w1       = (const float*)d_in[11];
  const float* n_b1       = (const float*)d_in[12];
  const float* n_w2       = (const float*)d_in[13];
  const float* n_b2       = (const float*)d_in[14];
  const float* ln_scale   = (const float*)d_in[15];
  const float* ln_bias    = (const float*)d_in[16];
  const float* dec_w      = (const float*)d_in[17];
  const float* dec_b      = (const float*)d_in[18];

  char* ws = (char*)d_ws;
  float* h    = (float*)(ws);                                      // 12.8 MB
  float* recv = (float*)(ws + (size_t)N_NODES * LATENT * 4);       // 12.8 MB
  unsigned short* e_buf = (unsigned short*)(ws + 2ull * N_NODES * LATENT * 4);  // 102.4 MB
  unsigned short* pw = (unsigned short*)(ws + 2ull * N_NODES * LATENT * 4
                                            + (size_t)N_EDGES * LATENT * 2);
  unsigned short* pw_e1[3];
  pw_e1[0] = pw;                       // 160x128
  pw_e1[1] = pw_e1[0] + 160 * 128;     // 224x128
  pw_e1[2] = pw_e1[1] + 224 * 128;     // 224x128
  unsigned short* pw_e2 = pw_e1[2] + 224 * 128;    // 3 x 128x64
  unsigned short* pw_n1 = pw_e2 + 3 * 128 * 64;    // 3 x 160x128
  unsigned short* pw_n2 = pw_n1 + 3 * 160 * 128;   // 3 x 128x64

  // pre-pack weights into WMMA B-fragment order (bf16)
  pack_w_kernel<<<(160 * 128 + 255) / 256, 256, 0, stream>>>(e_w1_first, pw_e1[0], 144, 160, 128);
  for (int t = 1; t < 3; ++t)
    pack_w_kernel<<<(224 * 128 + 255) / 256, 256, 0, stream>>>(
        e_w1_rest + (size_t)(t - 1) * 208 * 128, pw_e1[t], 208, 224, 128);
  for (int t = 0; t < 3; ++t)
    pack_w_kernel<<<(128 * 64 + 255) / 256, 256, 0, stream>>>(
        e_w2 + (size_t)t * 128 * 64, pw_e2 + t * 128 * 64, 128, 128, 64);
  for (int t = 0; t < 3; ++t)
    pack_w_kernel<<<(160 * 128 + 255) / 256, 256, 0, stream>>>(
        n_w1 + (size_t)t * 144 * 128, pw_n1 + t * 160 * 128, 144, 160, 128);
  for (int t = 0; t < 3; ++t)
    pack_w_kernel<<<(128 * 64 + 255) / 256, 256, 0, stream>>>(
        n_w2 + (size_t)t * 128 * 64, pw_n2 + t * 128 * 64, 128, 128, 64);

  embed_kernel<<<(N_NODES * LATENT + 255) / 256, 256, 0, stream>>>(nodes, embed_w, embed_b, h);

  for (int t = 0; t < 3; ++t) {
    zero_f32_kernel<<<(N_NODES * LATENT + 255) / 256, 256, 0, stream>>>(recv, N_NODES * LATENT);
    const int Kpad = (t == 0) ? 160 : 224;
    edge_mlp_kernel<<<N_EDGES / 32, 256, 0, stream>>>(
        h, e_buf, senders, receivers, g,
        pw_e1[t], e_b1 + t * 128, pw_e2 + t * 128 * 64, e_b2 + t * 64,
        recv, (t > 0) ? 1 : 0, Kpad);
    node_mlp_kernel<<<(N_NODES + 31) / 32, 256, 0, stream>>>(
        h, recv, g,
        pw_n1 + t * 160 * 128, n_b1 + t * 128, pw_n2 + t * 128 * 64, n_b2 + t * 64,
        ln_scale + t * 64, ln_bias + t * 64);
  }

  decode_kernel<<<(N_NODES * IN_F + 255) / 256, 256, 0, stream>>>(h, dec_w, dec_b, (float*)d_out);
}
